// DECSeq7_41180146434798
// MI455X (gfx1250) — matrix operand order repaired and checked by reading, not compiled, not run on hardware
//
#include <hip/hip_runtime.h>

#define NNODES 102400
#define NGRAPH 64
#define KDEG   16
#define NEDGE  (NNODES * KDEG)
#define RSPLIT 16

typedef unsigned short u16b;
typedef __bf16 bf16_t;
typedef bf16_t v16bf __attribute__((ext_vector_type(16)));
typedef float  v8f   __attribute__((ext_vector_type(8)));

union Frag { uint4 q[2]; u16b u[16]; v16bf v; };

__device__ __forceinline__ u16b f2bs(float f) {
  unsigned u = __float_as_uint(f);
  u += 0x7FFFu + ((u >> 16) & 1u);          // round-to-nearest-even bf16
  return (u16b)(u >> 16);
}
__device__ __forceinline__ float bs2f(u16b s) {
  return __uint_as_float(((unsigned)s) << 16);
}

// ---------------------------------------------------------------------------
// Utility: zero a float region (stats accumulators)
// ---------------------------------------------------------------------------
__global__ void zero_f32(float* __restrict__ p, int n) {
  int i = blockIdx.x * 256 + threadIdx.x;
  if (i < n) p[i] = 0.f;
}

// ---------------------------------------------------------------------------
// conv1 layer 1: per-edge [pos_i, pos_j - pos_i] (6) -> 64, relu, bf16 out,
// per-feature sum/sumsq stats via LDS reduction.
// ---------------------------------------------------------------------------
__global__ __launch_bounds__(256)
void conv1_layer1(const float* __restrict__ pos, const int* __restrict__ srcIdx,
                  const float* __restrict__ W, const float* __restrict__ b,
                  u16b* __restrict__ out, float* __restrict__ stats)
{
  __shared__ float sw[64 * 6];
  __shared__ float sb[64];
  __shared__ float ssum[64];
  __shared__ float ssq[64];
  int tid = threadIdx.x;
  for (int t = tid; t < 64 * 6; t += 256) sw[t] = W[t];
  if (tid < 64) { sb[tid] = b[tid]; ssum[tid] = 0.f; ssq[tid] = 0.f; }
  __syncthreads();

  size_t e = (size_t)blockIdx.x * 256 + tid;   // NEDGE divisible by 256
  int i = (int)(e >> 4);                       // tgt[e] = e/16 by construction
  int j = srcIdx[e];
  float m0 = pos[3 * i], m1 = pos[3 * i + 1], m2 = pos[3 * i + 2];
  float m3 = pos[3 * j] - m0, m4 = pos[3 * j + 1] - m1, m5 = pos[3 * j + 2] - m2;
  u16b* op = out + e * 64;
#pragma unroll 4
  for (int o = 0; o < 64; ++o) {
    const float* w = sw + o * 6;
    float acc = sb[o] + m0 * w[0] + m1 * w[1] + m2 * w[2]
                      + m3 * w[3] + m4 * w[4] + m5 * w[5];
    acc = acc > 0.f ? acc : 0.f;
    op[o] = f2bs(acc);
    atomicAdd(&ssum[o], acc);
    atomicAdd(&ssq[o], acc * acc);
  }
  __syncthreads();
  if (tid < 64) {
    atomicAdd(&stats[tid], ssum[tid]);
    atomicAdd(&stats[64 + tid], ssq[tid]);
  }
}

// ---------------------------------------------------------------------------
// Fold previous layer's BN (alpha,beta from stats) into this layer's weights:
//   W'[o,k] = bf16(W[o,k]*alpha[k]) ; b'[o] = b[o] + sum_k W[o,k]*beta[k]
// statsPrev == nullptr -> plain f32->bf16 conversion (alpha=1, beta=0).
// ---------------------------------------------------------------------------
__global__ __launch_bounds__(128)
void fold_w(const float* __restrict__ W, const float* __restrict__ b,
            const float* __restrict__ statsPrev, const float* __restrict__ gPrev,
            const float* __restrict__ bePrev, float invCnt,
            u16b* __restrict__ Wb, float* __restrict__ bOut, int K)
{
  int o = blockIdx.x;
  int tid = threadIdx.x;
  float accum = 0.f;
  for (int k = tid; k < K; k += 128) {
    float alpha = 1.f, beta = 0.f;
    if (statsPrev) {
      float mu  = statsPrev[k] * invCnt;
      float var = statsPrev[K + k] * invCnt - mu * mu;
      alpha = gPrev[k] * rsqrtf(var + 1e-5f);
      beta  = bePrev[k] - mu * alpha;
    }
    float wv = W[(size_t)o * K + k];
    Wb[(size_t)o * K + k] = f2bs(wv * alpha);
    accum += wv * beta;
  }
  __shared__ float red[128];
  red[tid] = accum;
  __syncthreads();
  for (int s = 64; s > 0; s >>= 1) {
    if (tid < s) red[tid] += red[tid + s];
    __syncthreads();
  }
  if (tid == 0) bOut[o] = b[o] + red[0];
}

// ---------------------------------------------------------------------------
// WMMA bf16 GEMM: C[rows, NOUT] = relu(A[rows,K] @ W[NOUT,K]^T + bias)
// One wave computes a 16 x (16*NB) output strip: NB accumulators share one
// A fragment per K-step (register blocking -> A read once per strip).
// A-operand source depends on MODE:
//   MODE 0: plain bf16 buffer A (row stride K)
//   MODE 1: fused conv2 gather: k<64 -> x1[i], k>=64 -> x1[src]-x1[i] (f32->bf16)
//   MODE 2: fused concat: k<64 -> xA (bf16 [rows,64]), else xB (bf16 [rows,128])
// Epilogue: bias + relu, bf16 store, per-feature sum/sumsq stats.
// ---------------------------------------------------------------------------
template<int MODE, int K, int NOUT, int NB>
__global__ __launch_bounds__(32)
void gemm_bf16(const u16b* __restrict__ A,
               const float* __restrict__ x1f, const int* __restrict__ srcIdx,
               const u16b* __restrict__ xA, const u16b* __restrict__ xB,
               const u16b* __restrict__ W, const float* __restrict__ bias,
               u16b* __restrict__ out, float* __restrict__ stats)
{
  const int lane = threadIdx.x & 31;
  const int r    = lane & 15;
  const int half = lane >> 4;
  const size_t rowTile = (size_t)blockIdx.x * 16;
  const size_t row = rowTile + r;
  const int colBase = blockIdx.y * (16 * NB);

  int gi = 0, gj = 0;
  if (MODE == 1) { gi = (int)(row >> 4); gj = srcIdx[row]; }

  v8f acc[NB];
#pragma unroll
  for (int nb = 0; nb < NB; ++nb)
    acc[nb] = (v8f){0.f, 0.f, 0.f, 0.f, 0.f, 0.f, 0.f, 0.f};

#pragma unroll
  for (int k0 = 0; k0 < K; k0 += 32) {
    Frag a;
    const int kb = k0 + half * 8;  // A layout: chunks [kb..kb+7], [kb+16..kb+23]
    if (MODE == 0) {
      const u16b* ar = A + row * K;
      a.q[0] = *(const uint4*)(ar + kb);
      a.q[1] = *(const uint4*)(ar + kb + 16);
    } else if (MODE == 1) {
#pragma unroll
      for (int c = 0; c < 2; ++c) {
        int kc = kb + c * 16;  // chunks never straddle the k=64 boundary
        if (kc < 64) {
          const float* p = x1f + (size_t)gi * 64 + kc;
#pragma unroll
          for (int t = 0; t < 8; ++t) a.u[c * 8 + t] = f2bs(p[t]);
        } else {
          const float* pi = x1f + (size_t)gi * 64 + (kc - 64);
          const float* pj = x1f + (size_t)gj * 64 + (kc - 64);
#pragma unroll
          for (int t = 0; t < 8; ++t) a.u[c * 8 + t] = f2bs(pj[t] - pi[t]);
        }
      }
    } else {
#pragma unroll
      for (int c = 0; c < 2; ++c) {
        int kc = kb + c * 16;
        const u16b* p = (kc < 64) ? (xA + row * 64 + kc)
                                  : (xB + row * 128 + (kc - 64));
        a.q[c] = *(const uint4*)p;
      }
    }
#pragma unroll
    for (int nb = 0; nb < NB; ++nb) {
      Frag b;
      // B operand: lane half h holds K chunks [k0+h*8..+7], [k0+16+h*8..+7] for its column
      const u16b* wr = W + (size_t)(colBase + nb * 16 + r) * K + (k0 + half * 8);
      b.q[0] = *(const uint4*)(wr);
      b.q[1] = *(const uint4*)(wr + 16);
      acc[nb] = __builtin_amdgcn_wmma_f32_16x16x32_bf16(
          false, a.v, false, b.v, (short)0, acc[nb], false, false);
    }
  }

#pragma unroll
  for (int nb = 0; nb < NB; ++nb) {
    const int col = colBase + nb * 16 + r;
    const float bv = bias[col];
    float s = 0.f, ss = 0.f;
#pragma unroll
    for (int jj = 0; jj < 8; ++jj) {
      float v = acc[nb][jj] + bv;
      v = v > 0.f ? v : 0.f;
      out[(rowTile + jj + 8 * half) * NOUT + col] = f2bs(v);
      s += v; ss += v * v;
    }
    // lanes (r, r+16) own the same column -> pairwise reduce, 1 atomic each
    s  += __shfl_down(s, 16, 32);
    ss += __shfl_down(ss, 16, 32);
    if (half == 0) {
      atomicAdd(&stats[col], s);
      atomicAdd(&stats[NOUT + col], ss);
    }
  }
}

// ---------------------------------------------------------------------------
// Per-node segment_max over KDEG edges fused with BN (alpha>0 -> commutes).
// Writes f32 (optional) and bf16 copies of x.
// ---------------------------------------------------------------------------
template<int F>
__global__ __launch_bounds__(256)
void segmax_bn_edge(const u16b* __restrict__ a, const float* __restrict__ stats,
                    const float* __restrict__ g, const float* __restrict__ be,
                    float invCnt, float* __restrict__ outF, u16b* __restrict__ outB)
{
  size_t idx = (size_t)blockIdx.x * 256 + threadIdx.x;
  if (idx >= (size_t)NNODES * F) return;
  int f = (int)(idx % F);
  size_t n = idx / F;
  const u16b* p = a + n * (size_t)KDEG * F + f;
  float m = 0.f;  // relu outputs are >= 0
#pragma unroll
  for (int e = 0; e < KDEG; ++e) m = fmaxf(m, bs2f(p[(size_t)e * F]));
  float mu  = stats[f] * invCnt;
  float var = stats[F + f] * invCnt - mu * mu;
  float v = (m - mu) * (g[f] * rsqrtf(var + 1e-5f)) + be[f];
  if (outF) outF[idx] = v;
  outB[idx] = f2bs(v);
}

// ---------------------------------------------------------------------------
// Graph max-pool stage 1: partial max over row chunks (RSPLIT-way parallel).
// ---------------------------------------------------------------------------
__global__ __launch_bounds__(256)
void segmax_batch_partial(const u16b* __restrict__ h, float* __restrict__ part)
{
  int idx = blockIdx.x * 256 + threadIdx.x;
  if (idx >= NGRAPH * 1024 * RSPLIT) return;
  int f  = idx & 1023;
  int t  = idx >> 10;
  int gr = t / RSPLIT;
  int ch = t % RSPLIT;
  const int rows  = NNODES / NGRAPH;   // 1600
  const int crows = rows / RSPLIT;     // 100
  const u16b* p = h + ((size_t)gr * rows + (size_t)ch * crows) * 1024 + f;
  float m = 0.f;
#pragma unroll 4
  for (int n = 0; n < crows; ++n) m = fmaxf(m, bs2f(p[(size_t)n * 1024]));
  part[idx] = m;
}

// ---------------------------------------------------------------------------
// Graph max-pool stage 2: reduce partials + BN of lin1.
// ---------------------------------------------------------------------------
__global__ __launch_bounds__(256)
void segmax_batch_final(const float* __restrict__ part, const float* __restrict__ stats,
                        const float* __restrict__ g, const float* __restrict__ be,
                        float invCnt, float* __restrict__ out)
{
  int idx = blockIdx.x * 256 + threadIdx.x;
  if (idx >= NGRAPH * 1024) return;
  int f  = idx & 1023;
  int gr = idx >> 10;
  float m = 0.f;
#pragma unroll
  for (int c = 0; c < RSPLIT; ++c)
    m = fmaxf(m, part[(((size_t)gr * RSPLIT + c) << 10) + f]);
  float mu  = stats[f] * invCnt;
  float var = stats[1024 + f] * invCnt - mu * mu;
  out[idx] = (m - mu) * (g[f] * rsqrtf(var + 1e-5f)) + be[f];
}

// ---------------------------------------------------------------------------
// Head: tiny scalar GEMM (G=64 rows) with optional relu + stats.
// ---------------------------------------------------------------------------
__global__ __launch_bounds__(256)
void head_gemm(const float* __restrict__ X, const float* __restrict__ W,
               const float* __restrict__ b, float* __restrict__ out,
               float* __restrict__ stats, int K, int NOUT, int doRelu)
{
  int idx = blockIdx.x * blockDim.x + threadIdx.x;
  if (idx >= NGRAPH * NOUT) return;
  int o  = idx % NOUT;
  int gr = idx / NOUT;
  const float* x = X + (size_t)gr * K;
  const float* w = W + (size_t)o * K;
  float acc = b[o];
  for (int k = 0; k < K; ++k) acc += x[k] * w[k];
  if (doRelu) acc = fmaxf(acc, 0.f);
  out[idx] = acc;
  if (stats) { atomicAdd(&stats[o], acc); atomicAdd(&stats[NOUT + o], acc * acc); }
}

__global__ __launch_bounds__(256)
void bn_apply(float* __restrict__ X, const float* __restrict__ stats,
              const float* __restrict__ g, const float* __restrict__ be,
              float invCnt, int F, int total)
{
  int idx = blockIdx.x * 256 + threadIdx.x;
  if (idx >= total) return;
  int f = idx % F;
  float mu  = stats[f] * invCnt;
  float var = stats[F + f] * invCnt - mu * mu;
  X[idx] = (X[idx] - mu) * (g[f] * rsqrtf(var + 1e-5f)) + be[f];
}

// ---------------------------------------------------------------------------
extern "C" void kernel_launch(void* const* d_in, const int* in_sizes, int n_in,
                              void* d_out, int out_size, void* d_ws, size_t ws_size,
                              hipStream_t stream)
{
  (void)in_sizes; (void)n_in; (void)out_size; (void)ws_size;
  const float* pos  = (const float*)d_in[0];
  const float* w1a  = (const float*)d_in[1];
  const float* b1a  = (const float*)d_in[2];
  const float* g1a  = (const float*)d_in[3];
  const float* be1a = (const float*)d_in[4];
  const float* w1b  = (const float*)d_in[5];
  const float* b1b  = (const float*)d_in[6];
  const float* g1b  = (const float*)d_in[7];
  const float* be1b = (const float*)d_in[8];
  const float* w1c  = (const float*)d_in[9];
  const float* b1c  = (const float*)d_in[10];
  const float* g1c  = (const float*)d_in[11];
  const float* be1c = (const float*)d_in[12];
  const float* w2   = (const float*)d_in[13];
  const float* b2   = (const float*)d_in[14];
  const float* g2   = (const float*)d_in[15];
  const float* be2  = (const float*)d_in[16];
  const float* wl   = (const float*)d_in[17];
  const float* bl   = (const float*)d_in[18];
  const float* gl   = (const float*)d_in[19];
  const float* bel  = (const float*)d_in[20];
  const float* wm1  = (const float*)d_in[21];
  const float* bm1  = (const float*)d_in[22];
  const float* gm1  = (const float*)d_in[23];
  const float* bem1 = (const float*)d_in[24];
  const float* wm2  = (const float*)d_in[25];
  const float* bm2  = (const float*)d_in[26];
  const float* gm2  = (const float*)d_in[27];
  const float* bem2 = (const float*)d_in[28];
  const float* wout = (const float*)d_in[29];
  const float* bout = (const float*)d_in[30];
  const int*   ei   = (const int*)d_in[31];
  const int* srcIdx = ei;   // edge_index row 0 = source nodes
  // batch (d_in[32]) not needed: nodes are contiguous, N/G per graph

  char* ws = (char*)d_ws;
  size_t off = 0;
  auto carve = [&](size_t bytes) -> char* {
    char* p = ws + off;
    off = (off + bytes + 255) & ~(size_t)255;
    return p;
  };

  float* statsAll = (float*)carve(4224 * sizeof(float));
  float* s1a = statsAll;         // 128
  float* s1b = statsAll + 128;   // 128
  float* s1c = statsAll + 256;   // 128
  float* s2  = statsAll + 384;   // 256
  float* sl  = statsAll + 640;   // 2048
  float* sm1 = statsAll + 2688;  // 1024
  float* sm2 = statsAll + 3712;  // 512

  u16b*  w1bF = (u16b*)carve(64 * 64 * 2);     float* b1bF = (float*)carve(64 * 4);
  u16b*  w1cF = (u16b*)carve(64 * 64 * 2);     float* b1cF = (float*)carve(64 * 4);
  u16b*  w2B  = (u16b*)carve(128 * 128 * 2);   float* b2F  = (float*)carve(128 * 4);
  u16b*  wlB  = (u16b*)carve(1024 * 192 * 2);  float* blF  = (float*)carve(1024 * 4);

  u16b*  a0   = (u16b*)carve((size_t)NEDGE * 64 * 2);
  u16b*  a1   = (u16b*)carve((size_t)NEDGE * 64 * 2);
  u16b*  a2   = (u16b*)carve((size_t)NEDGE * 128 * 2);
  float* x1f  = (float*)carve((size_t)NNODES * 64 * 4);
  u16b*  x1b  = (u16b*)carve((size_t)NNODES * 64 * 2);
  u16b*  x2b  = (u16b*)carve((size_t)NNODES * 128 * 2);
  u16b*  hN   = (u16b*)carve((size_t)NNODES * 1024 * 2);
  float* part = (float*)carve((size_t)NGRAPH * 1024 * RSPLIT * 4);
  float* y0   = (float*)carve(NGRAPH * 1024 * 4);
  float* t1   = (float*)carve(NGRAPH * 512 * 4);
  float* t2   = (float*)carve(NGRAPH * 256 * 4);

  const float invE = 1.f / (float)NEDGE;
  const float invN = 1.f / (float)NNODES;
  const float invG = 1.f / (float)NGRAPH;

  zero_f32<<<(4224 + 255) / 256, 256, 0, stream>>>(statsAll, 4224);

  // conv1: 6 -> 64 -> 64 -> 64 over edges, BN folded forward into weights
  conv1_layer1<<<NEDGE / 256, 256, 0, stream>>>(pos, srcIdx, w1a, b1a, a0, s1a);
  fold_w<<<64, 128, 0, stream>>>(w1b, b1b, s1a, g1a, be1a, invE, w1bF, b1bF, 64);
  gemm_bf16<0, 64, 64, 4><<<dim3(NEDGE / 16, 1), 32, 0, stream>>>(
      a0, nullptr, nullptr, nullptr, nullptr, w1bF, b1bF, a1, s1b);
  fold_w<<<64, 128, 0, stream>>>(w1c, b1c, s1b, g1b, be1b, invE, w1cF, b1cF, 64);
  gemm_bf16<0, 64, 64, 4><<<dim3(NEDGE / 16, 1), 32, 0, stream>>>(
      a1, nullptr, nullptr, nullptr, nullptr, w1cF, b1cF, a0, s1c);
  segmax_bn_edge<64><<<(NNODES * 64) / 256, 256, 0, stream>>>(
      a0, s1c, g1c, be1c, invE, x1f, x1b);

  // conv2: fused [x_i, x_j - x_i] gather -> 128x128 WMMA GEMM
  fold_w<<<128, 128, 0, stream>>>(w2, b2, nullptr, nullptr, nullptr, 0.f, w2B, b2F, 128);
  gemm_bf16<1, 128, 128, 4><<<dim3(NEDGE / 16, 2), 32, 0, stream>>>(
      nullptr, x1f, srcIdx, nullptr, nullptr, w2B, b2F, a2, s2);
  segmax_bn_edge<128><<<(NNODES * 128) / 256, 256, 0, stream>>>(
      a2, s2, g2, be2, invE, nullptr, x2b);

  // lin1: fused concat[x1,x2] -> 192x1024 WMMA GEMM, then graph max-pool + BN
  fold_w<<<1024, 128, 0, stream>>>(wl, bl, nullptr, nullptr, nullptr, 0.f, wlB, blF, 192);
  gemm_bf16<2, 192, 1024, 8><<<dim3(NNODES / 16, 8), 32, 0, stream>>>(
      nullptr, nullptr, nullptr, x1b, x2b, wlB, blF, hN, sl);
  segmax_batch_partial<<<(NGRAPH * 1024 * RSPLIT) / 256, 256, 0, stream>>>(hN, part);
  segmax_batch_final<<<(NGRAPH * 1024) / 256, 256, 0, stream>>>(part, sl, gl, bel, invN, y0);

  // head (G=64 rows, negligible FLOPs)
  head_gemm<<<(NGRAPH * 512 + 255) / 256, 256, 0, stream>>>(y0, wm1, bm1, t1, sm1, 1024, 512, 1);
  bn_apply<<<(NGRAPH * 512 + 255) / 256, 256, 0, stream>>>(t1, sm1, gm1, bem1, invG, 512, NGRAPH * 512);
  head_gemm<<<(NGRAPH * 256 + 255) / 256, 256, 0, stream>>>(t1, wm2, bm2, t2, sm2, 512, 256, 1);
  bn_apply<<<(NGRAPH * 256 + 255) / 256, 256, 0, stream>>>(t2, sm2, gm2, bem2, invG, 256, NGRAPH * 256);
  head_gemm<<<1, 128, 0, stream>>>(t2, wout, bout, (float*)d_out, nullptr, 256, 2, 0);
}
